// SparseMoE_82695300317516
// MI455X (gfx1250) — compile-verified, matrix operand order checked by reading, not run on hardware
//
#include <hip/hip_runtime.h>
#include <hip/hip_bf16.h>
#include <math.h>

// ---------------------------------------------------------------------------
// Problem constants (from reference): B=4, S=2048 -> T=8192 tokens
// ---------------------------------------------------------------------------
#define T_NUM 8192
#define H_DIM 1024
#define E_NUM 8
#define F_DIM 4096

// GEMM tiling: block computes BM x BN, stepping K by BK (one wmma k-step).
// 8 waves arranged 2(M) x 4(N); each wave owns a 64x64 tile = 16 WMMAs.
#define BM 128
#define BN 256
#define BK 32
#define KPAD 40   // halfs per LDS row: 80 bytes -> 16B aligned, bank-conflict-free

typedef __attribute__((ext_vector_type(16))) __bf16 v16bf;
typedef __attribute__((ext_vector_type(8)))  __bf16 v8bf;
typedef __attribute__((ext_vector_type(8)))  float  v8f;
typedef __attribute__((ext_vector_type(4)))  int    v4i;

// Address-space-qualified v4i for the async global->LDS builtin signature:
// (v4i addrspace(1)*, v4i addrspace(3)*, imm offset, imm cpol)
typedef __attribute__((address_space(1))) v4i v4i_g;
typedef __attribute__((address_space(3))) v4i v4i_l;

// CDNA5 async global->LDS path (ASYNCcnt) if the toolchain exposes it.
#if defined(__has_builtin)
#if __has_builtin(__builtin_amdgcn_global_load_async_to_lds_b128) && \
    __has_builtin(__builtin_amdgcn_s_wait_asynccnt)
#define HAVE_ASYNC_LDS 1
#endif
#endif
#ifndef HAVE_ASYNC_LDS
#define HAVE_ASYNC_LDS 0
#endif

__device__ __forceinline__ void copy16_to_lds(const unsigned short* g,
                                              unsigned short* l) {
#if HAVE_ASYNC_LDS
  __builtin_amdgcn_global_load_async_to_lds_b128((v4i_g*)g, (v4i_l*)l, 0, 0);
#else
  *reinterpret_cast<uint4*>(l) = *reinterpret_cast<const uint4*>(g);
#endif
}

__device__ __forceinline__ void stage_wait() {
#if HAVE_ASYNC_LDS
  __builtin_amdgcn_s_wait_asynccnt(0);
#endif
}

__device__ __forceinline__ unsigned short f2bf(float f) {
  unsigned u = __float_as_uint(f);
  u += 0x7FFFu + ((u >> 16) & 1u);          // round-to-nearest-even
  return (unsigned short)(u >> 16);
}

__device__ __forceinline__ float gelu_exact(float v) {
  return 0.5f * v * (1.0f + erff(v * 0.70710678118654752f));
}

// Build a v16bf A/B fragment for V_WMMA_*_16X16X32 from a padded LDS row.
// Per ISA 7.12.2 (16-bit A 16x32): lane<16 -> K=0..7 (elems 0-7), K=16..23
// (elems 8-15); lane>=16 -> K=8..15 and K=24..31.
__device__ __forceinline__ v16bf load_frag(const unsigned short* rowp, int half) {
  const v8bf lo = *reinterpret_cast<const v8bf*>(rowp + 8 * half);
  const v8bf hi = *reinterpret_cast<const v8bf*>(rowp + 16 + 8 * half);
  v16bf r;
#pragma unroll
  for (int i = 0; i < 8; ++i) { r[i] = lo[i]; r[i + 8] = hi[i]; }
  return r;
}

// ---------------------------------------------------------------------------
// 0) zero the output + per-expert counters (d_out/d_ws are poisoned)
// ---------------------------------------------------------------------------
__global__ void moe_init(float* __restrict__ out, int* __restrict__ counts) {
  size_t n = (size_t)T_NUM * H_DIM;
  for (size_t i = (size_t)blockIdx.x * blockDim.x + threadIdx.x; i < n;
       i += (size_t)gridDim.x * blockDim.x)
    out[i] = 0.0f;
  if (blockIdx.x == 0 && threadIdx.x < E_NUM) counts[threadIdx.x] = 0;
}

// ---------------------------------------------------------------------------
// 1) fp32 -> bf16 (activations)
// ---------------------------------------------------------------------------
__global__ void cvt_bf16(const float* __restrict__ in, unsigned short* __restrict__ outp,
                         size_t n) {
  for (size_t i = (size_t)blockIdx.x * blockDim.x + threadIdx.x; i < n;
       i += (size_t)gridDim.x * blockDim.x)
    outp[i] = f2bf(in[i]);
}

// fp32 -> bf16 with transpose: out[e][c][r] = in[e][r][c]  (writes coalesced)
__global__ void cvt_transpose_bf16(const float* __restrict__ in,
                                   unsigned short* __restrict__ outp,
                                   int R, int C) {
  size_t e = blockIdx.z;
  const float* ip = in + e * (size_t)R * C;
  unsigned short* op = outp + e * (size_t)R * C;
  size_t n = (size_t)R * C;
  for (size_t i = (size_t)blockIdx.x * blockDim.x + threadIdx.x; i < n;
       i += (size_t)gridDim.x * blockDim.x) {
    size_t c = i / (size_t)R;
    size_t r = i - c * (size_t)R;
    op[i] = f2bf(ip[r * (size_t)C + c]);
  }
}

// ---------------------------------------------------------------------------
// 2) gating: one wave32 per token. logits -> softmax -> top-2 -> renorm,
//    atomic append into per-expert (token, weight) lists.
// ---------------------------------------------------------------------------
__global__ void gate_topk(const float* __restrict__ x, const float* __restrict__ gw,
                          const float* __restrict__ gb, int* __restrict__ counts,
                          int* __restrict__ tok_list, float* __restrict__ w_list) {
  int wave_in_blk = threadIdx.x >> 5;
  int lane = threadIdx.x & 31;
  int t = blockIdx.x * (blockDim.x >> 5) + wave_in_blk;
  if (t >= T_NUM) return;

  const float* xr = x + (size_t)t * H_DIM;
  float acc[E_NUM];
#pragma unroll
  for (int e = 0; e < E_NUM; ++e) acc[e] = 0.0f;

  for (int h = lane; h < H_DIM; h += 32) {
    float xv = xr[h];
    const float* g = gw + (size_t)h * E_NUM;   // E contiguous per row
#pragma unroll
    for (int e = 0; e < E_NUM; ++e) acc[e] += xv * g[e];
  }
#pragma unroll
  for (int off = 16; off > 0; off >>= 1) {
#pragma unroll
    for (int e = 0; e < E_NUM; ++e) acc[e] += __shfl_xor(acc[e], off, 32);
  }

  if (lane == 0) {
    float mx = -3.4e38f;
#pragma unroll
    for (int e = 0; e < E_NUM; ++e) { acc[e] += gb[e]; mx = fmaxf(mx, acc[e]); }
    float p[E_NUM];
    float sum = 0.0f;
#pragma unroll
    for (int e = 0; e < E_NUM; ++e) { p[e] = __expf(acc[e] - mx); sum += p[e]; }
    int i0 = 0;
#pragma unroll
    for (int e = 1; e < E_NUM; ++e) if (p[e] > p[i0]) i0 = e;
    int i1 = (i0 == 0) ? 1 : 0;
#pragma unroll
    for (int e = 0; e < E_NUM; ++e) if (e != i0 && p[e] > p[i1]) i1 = e;
    float w0 = p[i0] / sum, w1 = p[i1] / sum;
    float s = fmaxf(w0 + w1, 1e-6f);
    w0 /= s; w1 /= s;
    int s0 = atomicAdd(&counts[i0], 1);
    tok_list[i0 * T_NUM + s0] = t;  w_list[i0 * T_NUM + s0] = w0;
    int s1 = atomicAdd(&counts[i1], 1);
    tok_list[i1 * T_NUM + s1] = t;  w_list[i1 * T_NUM + s1] = w1;
  }
}

// 3) prefix sum over 8 expert counts -> compact hidden-state row offsets
__global__ void scan_offsets(const int* __restrict__ counts, int* __restrict__ offsets) {
  if (threadIdx.x == 0) {
    int s = 0;
#pragma unroll
    for (int e = 0; e < E_NUM; ++e) { offsets[e] = s; s += counts[e]; }
  }
}

// ---------------------------------------------------------------------------
// 4) GEMM1 (gathered): hdn[slot, n] = gelu( x[tok(slot)] @ w1t[e]^T + b1[e] )
//    A: gathered tokens [cnt x H] bf16, B: w1t[e] stored [F][H] (K-contig).
//    Async double-buffered LDS staging; OOB rows clamped (their C rows are
//    discarded in the epilogue, so clamped garbage is harmless).
// ---------------------------------------------------------------------------
__launch_bounds__(256, 1)
__global__ void moe_gemm1(const unsigned short* __restrict__ xb,
                          const unsigned short* __restrict__ w1t,
                          const float* __restrict__ b1,
                          const int* __restrict__ counts,
                          const int* __restrict__ offsets,
                          const int* __restrict__ tok_list,
                          unsigned short* __restrict__ hdn) {
  int e = blockIdx.z;
  int cnt = counts[e];
  int m0 = blockIdx.y * BM;
  if (m0 >= cnt) return;
  int n0 = blockIdx.x * BN;

  __shared__ __align__(16) unsigned short As[2][BM][KPAD];
  __shared__ __align__(16) unsigned short Bs[2][BN][KPAD];
  __shared__ int s_tok[BM];

  int tid = threadIdx.x;
  if (tid < BM) {
    int slot = m0 + tid;
    int cl = slot < cnt ? slot : cnt - 1;          // clamp: always valid token
    s_tok[tid] = tok_list[e * T_NUM + cl];
  }
  __syncthreads();

  const unsigned short* Bg = w1t + ((size_t)e * F_DIM + n0) * H_DIM;

  auto stage = [&](int sb, int kk) {
#pragma unroll
    for (int c = 0; c < 2; ++c) {                  // A: 512 16B chunks
      int chunk = tid + c * 256;
      int r = chunk >> 2;
      int kc = (chunk & 3) * 8;
      copy16_to_lds(xb + (size_t)s_tok[r] * H_DIM + kk + kc, &As[sb][r][kc]);
    }
#pragma unroll
    for (int c = 0; c < 4; ++c) {                  // B: 1024 16B chunks
      int chunk = tid + c * 256;
      int r = chunk >> 2;
      int kc = (chunk & 3) * 8;
      copy16_to_lds(Bg + (size_t)r * H_DIM + kk + kc, &Bs[sb][r][kc]);
    }
  };

  int wave = tid >> 5, lane = tid & 31;
  int wm = (wave & 1) * 64;        // 2 waves along M
  int wn = (wave >> 1) * 64;       // 4 waves along N
  int half = lane >> 4, l16 = lane & 15;

  v8f acc[4][4];
#pragma unroll
  for (int i = 0; i < 4; ++i)
#pragma unroll
    for (int j = 0; j < 4; ++j)
#pragma unroll
      for (int q = 0; q < 8; ++q) acc[i][j][q] = 0.0f;

  stage(0, 0);
  stage_wait();
  __syncthreads();

  int buf = 0;
  for (int k0 = 0; k0 < H_DIM; k0 += BK) {
    if (k0 + BK < H_DIM) stage(buf ^ 1, k0 + BK);  // overlap with compute

    v16bf a[4], b[4];
#pragma unroll
    for (int i = 0; i < 4; ++i) a[i] = load_frag(&As[buf][wm + i * 16 + l16][0], half);
#pragma unroll
    for (int j = 0; j < 4; ++j) b[j] = load_frag(&Bs[buf][wn + j * 16 + l16][0], half);

#pragma unroll
    for (int i = 0; i < 4; ++i)
#pragma unroll
      for (int j = 0; j < 4; ++j)
        acc[i][j] = __builtin_amdgcn_wmma_f32_16x16x32_bf16(
            false, a[i], false, b[j], (short)0, acc[i][j], false, false);

    stage_wait();
    __syncthreads();
    buf ^= 1;
  }

  // epilogue: +b1, gelu, store bf16 into compact hidden buffer
  int base_row = offsets[e] + m0;
#pragma unroll
  for (int i = 0; i < 4; ++i) {
#pragma unroll
    for (int j = 0; j < 4; ++j) {
      int col = n0 + wn + j * 16 + l16;
      float bias = b1[(size_t)e * F_DIM + col];
#pragma unroll
      for (int r = 0; r < 8; ++r) {
        int row = wm + i * 16 + r + 8 * half;
        if (m0 + row < cnt) {
          float v = gelu_exact(acc[i][j][r] + bias);
          hdn[(size_t)(base_row + row) * F_DIM + col] = f2bf(v);
        }
      }
    }
  }
}

// ---------------------------------------------------------------------------
// 5) GEMM2 (scatter): out[tok] += w * ( hdn[slot] @ w2t[e]^T + b2[e] )
//    A: hdn [cnt x F] bf16, B: w2t[e] stored [H][F] (K-contig).
// ---------------------------------------------------------------------------
__launch_bounds__(256, 1)
__global__ void moe_gemm2(const unsigned short* __restrict__ hdn,
                          const unsigned short* __restrict__ w2t,
                          const float* __restrict__ b2,
                          const int* __restrict__ counts,
                          const int* __restrict__ offsets,
                          const int* __restrict__ tok_list,
                          const float* __restrict__ w_list,
                          float* __restrict__ out) {
  int e = blockIdx.z;
  int cnt = counts[e];
  int m0 = blockIdx.y * BM;
  if (m0 >= cnt) return;
  int n0 = blockIdx.x * BN;

  __shared__ __align__(16) unsigned short As[2][BM][KPAD];
  __shared__ __align__(16) unsigned short Bs[2][BN][KPAD];
  __shared__ int s_tok[BM];
  __shared__ float s_w[BM];

  int tid = threadIdx.x;
  if (tid < BM) {
    int slot = m0 + tid;
    int cl = slot < cnt ? slot : cnt - 1;          // clamp: always valid slot
    s_tok[tid] = tok_list[e * T_NUM + cl];
    s_w[tid]   = w_list[e * T_NUM + cl];
  }
  __syncthreads();

  const unsigned short* Ab = hdn + (size_t)offsets[e] * F_DIM;
  const unsigned short* Bg = w2t + ((size_t)e * H_DIM + n0) * F_DIM;
  int row_hi = cnt - 1;

  auto stage = [&](int sb, int kk) {
#pragma unroll
    for (int c = 0; c < 2; ++c) {
      int chunk = tid + c * 256;
      int r = chunk >> 2;
      int kc = (chunk & 3) * 8;
      int gr = m0 + r; gr = gr < row_hi ? gr : row_hi;  // clamp to valid rows
      copy16_to_lds(Ab + (size_t)gr * F_DIM + kk + kc, &As[sb][r][kc]);
    }
#pragma unroll
    for (int c = 0; c < 4; ++c) {
      int chunk = tid + c * 256;
      int r = chunk >> 2;
      int kc = (chunk & 3) * 8;
      copy16_to_lds(Bg + (size_t)r * F_DIM + kk + kc, &Bs[sb][r][kc]);
    }
  };

  int wave = tid >> 5, lane = tid & 31;
  int wm = (wave & 1) * 64;
  int wn = (wave >> 1) * 64;
  int half = lane >> 4, l16 = lane & 15;

  v8f acc[4][4];
#pragma unroll
  for (int i = 0; i < 4; ++i)
#pragma unroll
    for (int j = 0; j < 4; ++j)
#pragma unroll
      for (int q = 0; q < 8; ++q) acc[i][j][q] = 0.0f;

  stage(0, 0);
  stage_wait();
  __syncthreads();

  int buf = 0;
  for (int k0 = 0; k0 < F_DIM; k0 += BK) {
    if (k0 + BK < F_DIM) stage(buf ^ 1, k0 + BK);

    v16bf a[4], b[4];
#pragma unroll
    for (int i = 0; i < 4; ++i) a[i] = load_frag(&As[buf][wm + i * 16 + l16][0], half);
#pragma unroll
    for (int j = 0; j < 4; ++j) b[j] = load_frag(&Bs[buf][wn + j * 16 + l16][0], half);

#pragma unroll
    for (int i = 0; i < 4; ++i)
#pragma unroll
      for (int j = 0; j < 4; ++j)
        acc[i][j] = __builtin_amdgcn_wmma_f32_16x16x32_bf16(
            false, a[i], false, b[j], (short)0, acc[i][j], false, false);

    stage_wait();
    __syncthreads();
    buf ^= 1;
  }

  // epilogue: +b2, scale by routing weight, atomic scatter-add into out[token]
#pragma unroll
  for (int i = 0; i < 4; ++i) {
#pragma unroll
    for (int j = 0; j < 4; ++j) {
      int col = n0 + wn + j * 16 + l16;
      float bias = b2[(size_t)e * H_DIM + col];
#pragma unroll
      for (int r = 0; r < 8; ++r) {
        int row = wm + i * 16 + r + 8 * half;
        if (m0 + row < cnt) {
          float v = (acc[i][j][r] + bias) * s_w[row];
          atomicAdd(&out[(size_t)s_tok[row] * H_DIM + col], v);
        }
      }
    }
  }
}

// ---------------------------------------------------------------------------
// host launcher
// ---------------------------------------------------------------------------
extern "C" void kernel_launch(void* const* d_in, const int* in_sizes, int n_in,
                              void* d_out, int out_size, void* d_ws, size_t ws_size,
                              hipStream_t stream) {
  (void)in_sizes; (void)n_in; (void)out_size; (void)ws_size;
  const float* x  = (const float*)d_in[0];
  const float* gw = (const float*)d_in[1];
  const float* gb = (const float*)d_in[2];
  const float* w1 = (const float*)d_in[3];
  const float* b1 = (const float*)d_in[4];
  const float* w2 = (const float*)d_in[5];
  const float* b2 = (const float*)d_in[6];
  float* out = (float*)d_out;

  char* ws = (char*)d_ws;
  size_t off = 0;
  auto alloc = [&](size_t bytes) -> char* {
    off = (off + 255) & ~(size_t)255;
    char* p = ws + off;
    off += bytes;
    return p;
  };

  int*            counts   = (int*)  alloc(E_NUM * sizeof(int));
  int*            offsets  = (int*)  alloc(E_NUM * sizeof(int));
  int*            tok_list = (int*)  alloc((size_t)E_NUM * T_NUM * sizeof(int));
  float*          w_list   = (float*)alloc((size_t)E_NUM * T_NUM * sizeof(float));
  unsigned short* xb       = (unsigned short*)alloc((size_t)T_NUM * H_DIM * 2);
  unsigned short* w1t      = (unsigned short*)alloc((size_t)E_NUM * H_DIM * F_DIM * 2);
  unsigned short* w2t      = (unsigned short*)alloc((size_t)E_NUM * F_DIM * H_DIM * 2);
  unsigned short* hdn      = (unsigned short*)alloc((size_t)T_NUM * 2 * F_DIM * 2);

  moe_init<<<512, 256, 0, stream>>>(out, counts);
  cvt_bf16<<<2048, 256, 0, stream>>>(x, xb, (size_t)T_NUM * H_DIM);
  cvt_transpose_bf16<<<dim3(2048, 1, E_NUM), 256, 0, stream>>>(w1, w1t, H_DIM, F_DIM);
  cvt_transpose_bf16<<<dim3(2048, 1, E_NUM), 256, 0, stream>>>(w2, w2t, F_DIM, H_DIM);
  gate_topk<<<T_NUM / 8, 256, 0, stream>>>(x, gw, gb, counts, tok_list, w_list);
  scan_offsets<<<1, 32, 0, stream>>>(counts, offsets);
  moe_gemm1<<<dim3(F_DIM / BN, T_NUM / BM, E_NUM), 256, 0, stream>>>(
      xb, w1t, b1, counts, offsets, tok_list, hdn);
  moe_gemm2<<<dim3(H_DIM / BN, T_NUM / BM, E_NUM), 256, 0, stream>>>(
      hdn, w2t, b2, counts, offsets, tok_list, w_list, out);
}
